// QLSTM_65481071406037
// MI455X (gfx1250) — matrix operand order repaired and checked by reading, not compiled, not run on hardware
//
#include <hip/hip_runtime.h>

// ---------------------------------------------------------------------------
// LSTM on MI455X (gfx1250, wave32, WMMA).
//
//  * fp32 emulated via bf16 hi/lo split: 3x v_wmma_f32_16x16x32_bf16 per
//    K=32 tile (vs native f32 WMMA's K=4) at near-fp32 accuracy.
//  * Activations stored as SEPARATE hi / lo bf16-pair planes in exact
//    A-fragment order -> every A operand is two direct global_load_b128,
//    no register shuffling (kills the v_dual_mov + v_nop hazard chains).
//  * Weight slab in LDS, column-major padded (stride 516 dwords: 16 cols
//    hit all 64 banks, conflict-free ds_load_b128). Per-gate base pointers
//    hoisted; K-loops fully unrolled -> all LDS/global offsets immediate.
//  * 32 persistent WGs; each owns 16 hidden cols (N-tile g == gate g).
//    h double-buffered in global (L2-resident); atomic grid barrier per
//    step; cell state c in registers for the whole sequence.
// ---------------------------------------------------------------------------

typedef __attribute__((ext_vector_type(16))) __bf16 v16bf;
typedef __attribute__((ext_vector_type(8)))  float  v8f;

#define T_STEPS 512
#define BATCH   128
#define NIN     512
#define NH      512
#define DTOT    1024            // NIN + NH
#define NWG     32              // persistent workgroups
#define SLICE   16              // hidden columns per workgroup
#define GCOLS   2048            // 4 gates * NH
#define KPAIRS  (DTOT / 2)      // 512 bf16-pair rows in the slab
#define SSTRIDE 516             // padded LDS col stride (dwords): 516%64==4

union BV { v16bf v; unsigned u[8]; uint4 q[2]; };

__device__ __forceinline__ float sigmoidf_(float x) {
    return 1.0f / (1.0f + __expf(-x));
}
__device__ __forceinline__ float tanhf_(float x) {
    return 2.0f / (1.0f + __expf(-2.0f * x)) - 1.0f;
}

// Pack two bf16(hi) of (a,b) into one dword (v_cvt_pk_bf16_f32 friendly).
__device__ __forceinline__ unsigned pk2(float a, float b) {
    __bf16 ha = (__bf16)a, hb = (__bf16)b;
    return (unsigned)__builtin_bit_cast(unsigned short, ha)
         | ((unsigned)__builtin_bit_cast(unsigned short, hb) << 16);
}

// ---------------------------------------------------------------------------
// Weight prep: W_g[n][k] fp32 -> Whi/Wlo bf16 bits, layout [k][g*512+n].
// ---------------------------------------------------------------------------
__global__ void prep_weights(const float* __restrict__ Wf, const float* __restrict__ Wi,
                             const float* __restrict__ Wg, const float* __restrict__ Wo,
                             unsigned short* __restrict__ Whi,
                             unsigned short* __restrict__ Wlo) {
    int idx = blockIdx.x * blockDim.x + threadIdx.x;     // over 4*512*1024
    if (idx >= 4 * NH * DTOT) return;
    int g   = idx >> 19;
    int rem = idx & (NH * DTOT - 1);
    int n   = rem >> 10;
    int k   = rem & (DTOT - 1);
    const float* W = (g == 0) ? Wf : (g == 1) ? Wi : (g == 2) ? Wg : Wo;
    float w = W[(size_t)n * DTOT + k];
    __bf16 hi = (__bf16)w;
    __bf16 lo = (__bf16)(w - (float)hi);
    size_t dst = (size_t)k * GCOLS + (size_t)g * NH + n;
    Whi[dst] = __builtin_bit_cast(unsigned short, hi);
    Wlo[dst] = __builtin_bit_cast(unsigned short, lo);
}

// ---------------------------------------------------------------------------
// x prep: split x[t,b,k] fp32 into two bf16-pair planes (hi / lo), each in
// A-fragment pair order. Fully parallel; no conversion in the hot loop.
// ---------------------------------------------------------------------------
__global__ void prep_x(const float* __restrict__ x,
                       unsigned* __restrict__ xHi, unsigned* __restrict__ xLo) {
    size_t i = (size_t)blockIdx.x * blockDim.x + threadIdx.x;  // pair index
    if (i >= (size_t)T_STEPS * BATCH * (NIN / 2)) return;
    float2 v = ((const float2*)x)[i];
    __bf16 ha = (__bf16)v.x, hb = (__bf16)v.y;
    float la = v.x - (float)ha, lb = v.y - (float)hb;
    xHi[i] = pk2(v.x, v.y);
    xLo[i] = pk2(la, lb);
}

// Zero h0 planes (0x0000 == bf16 zero) and the barrier counter.
__global__ void init_state(unsigned* __restrict__ hHi, unsigned* __restrict__ hLo,
                           unsigned* __restrict__ cnt) {
    int i = blockIdx.x * blockDim.x + threadIdx.x;
    if (i < BATCH * (NH / 2)) { hHi[i] = 0u; hLo[i] = 0u; }
    if (i == 0) *cnt = 0u;
}

// ---------------------------------------------------------------------------
// Persistent LSTM: grid = 32 WGs x 256 threads (8 waves).
// Wave w: batch rows [16w,16w+16). WG b: hidden cols [16b,16b+16).
// ---------------------------------------------------------------------------
__global__ __launch_bounds__(256, 1)
void lstm_persistent(const unsigned* __restrict__ xHi,   // [T*B][256] hi pairs
                     const unsigned* __restrict__ xLo,   // [T*B][256] lo pairs
                     const float* __restrict__ bfv, const float* __restrict__ biv,
                     const float* __restrict__ bgv, const float* __restrict__ bov,
                     const unsigned short* __restrict__ Whi,
                     const unsigned short* __restrict__ Wlo,
                     unsigned* __restrict__ hHi,         // [2][B][256] hi pairs
                     unsigned* __restrict__ hLo,         // [2][B][256] lo pairs
                     unsigned* __restrict__ cnt,
                     float* __restrict__ out) {          // outputs | hx | cx
    // Weight slab, column-major padded: slab[col*516 + kpair], 2x129 KB.
    __shared__ unsigned slabHi[64 * SSTRIDE];
    __shared__ unsigned slabLo[64 * SSTRIDE];

    const int tid  = threadIdx.x;
    const int wg   = blockIdx.x;          // 0..31
    const int lane = tid & 31;
    const int wave = tid >> 5;            // 0..7
    const int lrow = lane & 15;
    const int half = lane >> 4;
    const int m0   = wave * 16;

    // ---- stage weight slab (pairs packed: low u16 = even k) ----
    for (int i = tid; i < KPAIRS * 64; i += 256) {
        int kk  = i >> 6;                 // pair index 0..511
        int col = i & 63;                 // local gate-col
        int g = col >> 4, j = col & 15;
        size_t gcol = (size_t)g * NH + wg * SLICE + j;
        unsigned short a0 = Whi[(size_t)(2 * kk)     * GCOLS + gcol];
        unsigned short a1 = Whi[(size_t)(2 * kk + 1) * GCOLS + gcol];
        slabHi[col * SSTRIDE + kk] = (unsigned)a0 | ((unsigned)a1 << 16);
        unsigned short c0 = Wlo[(size_t)(2 * kk)     * GCOLS + gcol];
        unsigned short c1 = Wlo[(size_t)(2 * kk + 1) * GCOLS + gcol];
        slabLo[col * SSTRIDE + kk] = (unsigned)c0 | ((unsigned)c1 << 16);
    }
    __syncthreads();

    // Hoisted per-gate LDS base pointers (col*516 + half*4 folded in).
    const unsigned* sbH[4];
    const unsigned* sbL[4];
#pragma unroll
    for (int g = 0; g < 4; ++g) {
        const int col = g * 16 + lrow;
        sbH[g] = &slabHi[col * SSTRIDE + half * 4];
        sbL[g] = &slabLo[col * SSTRIDE + half * 4];
    }

    const int hcol = wg * SLICE + lrow;   // this lane's C/D column
    float bias[4];
    bias[0] = bfv[hcol]; bias[1] = biv[hcol]; bias[2] = bgv[hcol]; bias[3] = bov[hcol];

    float creg[8];
#pragma unroll
    for (int r = 0; r < 8; ++r) creg[r] = 0.0f;

    const int arow = m0 + lrow;           // A-matrix row (batch) for loads

    for (int t = 0; t < T_STEPS; ++t) {
        // Per-region A bases (half*4 pair offset folded in).
        const unsigned* xh = xHi + ((size_t)t * BATCH + arow) * (NIN / 2) + half * 4;
        const unsigned* xl = xLo + ((size_t)t * BATCH + arow) * (NIN / 2) + half * 4;
        const unsigned* hh = hHi + (size_t)(t & 1) * (BATCH * NH / 2)
                                 + (size_t)arow * (NH / 2) + half * 4;
        const unsigned* hl = hLo + (size_t)(t & 1) * (BATCH * NH / 2)
                                 + (size_t)arow * (NH / 2) + half * 4;
        unsigned* hhw = hHi + (size_t)((t + 1) & 1) * (BATCH * NH / 2);
        unsigned* hlw = hLo + (size_t)((t + 1) & 1) * (BATCH * NH / 2);

        v8f acc[4];
#pragma unroll
        for (int g = 0; g < 4; ++g)
#pragma unroll
            for (int r = 0; r < 8; ++r) acc[g][r] = bias[g];

        // One K=512 region (16 K-tiles), fully unrolled: immediate offsets only.
        auto kregion = [&](const unsigned* aH, const unsigned* aL, int slabBase) {
#pragma unroll
            for (int kt = 0; kt < 16; ++kt) {
                const int kp = kt * 16;                  // pair offset in region

                // ---- A fragment: two b128 per plane, lands directly ----
                BV ahi, alo;
                ahi.q[0] = *(const uint4*)(aH + kp);
                ahi.q[1] = *(const uint4*)(aH + kp + 8);
                alo.q[0] = *(const uint4*)(aL + kp);
                alo.q[1] = *(const uint4*)(aL + kp + 8);

                const int sk = slabBase + kp;            // immediate LDS offset
#pragma unroll
                for (int g = 0; g < 4; ++g) {
                    BV bhi, blo;
                    bhi.q[0] = *(const uint4*)(sbH[g] + sk);
                    bhi.q[1] = *(const uint4*)(sbH[g] + sk + 8);
                    blo.q[0] = *(const uint4*)(sbL[g] + sk);
                    blo.q[1] = *(const uint4*)(sbL[g] + sk + 8);
                    acc[g] = __builtin_amdgcn_wmma_f32_16x16x32_bf16(
                        false, ahi.v, false, bhi.v, (short)0, acc[g], false, false);
                    acc[g] = __builtin_amdgcn_wmma_f32_16x16x32_bf16(
                        false, ahi.v, false, blo.v, (short)0, acc[g], false, false);
                    acc[g] = __builtin_amdgcn_wmma_f32_16x16x32_bf16(
                        false, alo.v, false, bhi.v, (short)0, acc[g], false, false);
                }
            }
        };
        kregion(xh, xl, 0);               // K = 0..511   (x part)
        kregion(hh, hl, KPAIRS / 2);      // K = 512..1023 (h part)

        // ---- gates + state update; emit fp32 h and split-bf16 planes ----
#pragma unroll
        for (int r = 0; r < 8; ++r) {
            float fg = sigmoidf_(acc[0][r]);
            float ig = sigmoidf_(acc[1][r]);
            float gg = tanhf_(acc[2][r]);
            float og = sigmoidf_(acc[3][r]);
            float c  = fg * creg[r] + ig * gg;
            float h  = og * tanhf_(c);
            creg[r]  = c;
            const int row = m0 + half * 8 + r;
            out[((size_t)t * BATCH + row) * NH + hcol] = h;
            if (t == T_STEPS - 1)
                out[(size_t)T_STEPS * BATCH * NH + (size_t)row * NH + hcol] = h;  // hx
            // Pair adjacent columns; even lane stores the packed planes.
            float hn = __shfl_xor(h, 1, 32);
            if ((lrow & 1) == 0) {
                __bf16 h0 = (__bf16)h, h1 = (__bf16)hn;
                float l0 = h - (float)h0, l1 = hn - (float)h1;
                size_t pi = (size_t)row * (NH / 2) + (hcol >> 1);
                hhw[pi] = pk2(h, hn);
                hlw[pi] = pk2(l0, l1);
            }
        }

        // ---- grid-wide release/acquire barrier before next step ----
        __syncthreads();
        if (tid == 0) {
            __threadfence();
            atomicAdd(cnt, 1u);
            unsigned target = (unsigned)(t + 1) * (unsigned)gridDim.x;
            while (atomicAdd(cnt, 0u) < target)
                __builtin_amdgcn_s_sleep(1);
            __threadfence();
        }
        __syncthreads();
    }

    // ---- final cell state cx ----
#pragma unroll
    for (int r = 0; r < 8; ++r) {
        int row = m0 + half * 8 + r;
        out[(size_t)T_STEPS * BATCH * NH + (size_t)BATCH * NH
            + (size_t)row * NH + hcol] = creg[r];
    }
}

// ---------------------------------------------------------------------------
extern "C" void kernel_launch(void* const* d_in, const int* in_sizes, int n_in,
                              void* d_out, int out_size, void* d_ws, size_t ws_size,
                              hipStream_t stream) {
    (void)in_sizes; (void)n_in; (void)out_size; (void)ws_size;
    const float* x  = (const float*)d_in[0];
    const float* Wf = (const float*)d_in[1];
    const float* bf = (const float*)d_in[2];
    const float* Wi = (const float*)d_in[3];
    const float* bi = (const float*)d_in[4];
    const float* Wg = (const float*)d_in[5];
    const float* bg = (const float*)d_in[6];
    const float* Wo = (const float*)d_in[7];
    const float* bo = (const float*)d_in[8];

    // Workspace: Whi 4MB | Wlo 4MB | xHi 67MB | xLo 67MB | hHi 256KB | hLo 256KB | cnt
    char* ws = (char*)d_ws;
    const size_t szW  = (size_t)DTOT * GCOLS * 2;          // 4 MB each
    const size_t szXp = (size_t)T_STEPS * BATCH * (NIN / 2) * 4;  // 67 MB each
    const size_t szHp = (size_t)2 * BATCH * (NH / 2) * 4;  // 256 KB each
    unsigned short* Whi = (unsigned short*)ws;
    unsigned short* Wlo = (unsigned short*)(ws + szW);
    unsigned* xHi = (unsigned*)(ws + 2 * szW);
    unsigned* xLo = (unsigned*)(ws + 2 * szW + szXp);
    unsigned* hHi = (unsigned*)(ws + 2 * szW + 2 * szXp);
    unsigned* hLo = (unsigned*)(ws + 2 * szW + 2 * szXp + szHp);
    unsigned* cnt = (unsigned*)(ws + 2 * szW + 2 * szXp + 2 * szHp);

    prep_weights<<<(4 * NH * DTOT + 255) / 256, 256, 0, stream>>>(Wf, Wi, Wg, Wo, Whi, Wlo);
    prep_x<<<(T_STEPS * BATCH * (NIN / 2) + 255) / 256, 256, 0, stream>>>(x, xHi, xLo);
    init_state<<<(BATCH * (NH / 2) + 255) / 256, 256, 0, stream>>>(hHi, hLo, cnt);
    lstm_persistent<<<NWG, 256, 0, stream>>>(xHi, xLo, bf, bi, bg, bo, Whi, Wlo,
                                             hHi, hLo, cnt, (float*)d_out);
}
